// BigBirdEncoder_41669772705974
// MI455X (gfx1250) — matrix operand order repaired and checked by reading, not compiled, not run on hardware
//
#include <hip/hip_runtime.h>
#include <hip/hip_bf16.h>

typedef __bf16 bf16;
typedef __attribute__((ext_vector_type(16))) __bf16 v16bf;
typedef __attribute__((ext_vector_type(8)))  __bf16 v8bf;
typedef __attribute__((ext_vector_type(8)))  float  v8f;

// Problem constants (from reference)
constexpr int Bn = 8, Sn = 2048, Dn = 256, Fn = 1024, Hn = 8, DHn = 32;
constexpr int NBn = 32, BSn = 64, Mn = 7, MKn = 448; // 7 blocks * 64 keys

// ---------------------------------------------------------------------------
// Weight transpose + fp32 -> bf16 : out[n*K + k] = (bf16) w[k*N + n]
// ---------------------------------------------------------------------------
__global__ void wtrans_k(const float* __restrict__ w, bf16* __restrict__ out,
                         int K, int N) {
  int id = blockIdx.x * 256 + threadIdx.x;
  if (id >= K * N) return;
  int n = id / K, k = id % K;
  out[id] = (bf16)w[(size_t)k * N + n];
}

// ---------------------------------------------------------------------------
// Embedding gather: x[row,:] = emb[ids[row],:]
// ---------------------------------------------------------------------------
__global__ void embed_k(const long long* __restrict__ ids,
                        const float* __restrict__ emb,
                        float* __restrict__ x) {
  int row = blockIdx.x;
  long long id = ids[row];
  x[(size_t)row * Dn + threadIdx.x] = emb[(size_t)id * Dn + threadIdx.x];
}

// ---------------------------------------------------------------------------
// RMSNorm: one wave (32 lanes) per row of 256; lane holds 8 elems.
// ---------------------------------------------------------------------------
template <typename OUT>
__global__ __launch_bounds__(256) void rmsnorm_k(const float* __restrict__ x,
                                                 const float* __restrict__ w,
                                                 OUT* __restrict__ out, int rows) {
  int lane = threadIdx.x & 31;
  int wave = threadIdx.x >> 5;
  int row  = blockIdx.x * 8 + wave;
  if (row >= rows) return;
  const float* xr = x + (size_t)row * Dn + lane * 8;
  float v[8]; float ss = 0.f;
#pragma unroll
  for (int i = 0; i < 8; i++) { v[i] = xr[i]; ss += v[i] * v[i]; }
#pragma unroll
  for (int o = 16; o > 0; o >>= 1) ss += __shfl_xor(ss, o, 32);
  float r = rsqrtf(ss * (1.0f / 256.0f) + 1e-8f);
  OUT* orow = out + (size_t)row * Dn + lane * 8;
  const float* wr = w + lane * 8;
#pragma unroll
  for (int i = 0; i < 8; i++) orow[i] = (OUT)(v[i] * r * wr[i]);
}

// ---------------------------------------------------------------------------
// bf16 WMMA GEMM: C[M,N] = A[M,K] @ BT[N,K]^T  (BT = weight transposed)
// Each wave: one 32x64 tile = 2 row-tiles x 4 col-tiles (8 accumulators);
// every B load feeds two WMMAs. All B tiles loaded into distinct registers
// before the WMMA burst so the compiler can clause loads + stagger waits.
// Epilogue modes:
//   0: q  -> bf16 [b,h,s,d] with 1/sqrt(dh) scale
//   1: k  -> bf16 [b,h,s,d]
//   2: v  -> bf16 [b,h,d,s]  (transposed)
//   3: residual: outf[m*N+c] += acc   (fp32)
//   4: gelu -> bf16 [m*N+c]
// ---------------------------------------------------------------------------
__global__ __launch_bounds__(256) void gemm_bf16_k(
    const bf16* __restrict__ A, const bf16* __restrict__ BT,
    int M, int K, int N, int mode,
    bf16* __restrict__ outb, float* __restrict__ outf) {
  const int lane = threadIdx.x & 31;
  const int wave = threadIdx.x >> 5;
  const int lr = lane & 15, lh = lane >> 4;
  const int cgs = N >> 6;                       // 64-col groups
  int job = blockIdx.x * 8 + wave;
  int totalJobs = (M >> 5) * cgs;               // 32-row tiles
  if (job >= totalJobs) return;
  int rowTile = job / cgs, cg = job % cgs;
  int rowBase = rowTile << 5;
  const bf16* aRow0 = A + (size_t)(rowBase + lr) * K;
  const bf16* aRow1 = A + (size_t)(rowBase + 16 + lr) * K;
  const int kpart = lh << 4;                    // 0 or 16
  v8f acc[2][4];
#pragma unroll
  for (int rt = 0; rt < 2; rt++)
#pragma unroll
    for (int ct = 0; ct < 4; ct++) acc[rt][ct] = (v8f){};

  for (int kk = 0; kk < K; kk += 32) {
    // A operands: contiguous k = 16*h + s (two 8-elem chunks)
    v16bf a0, a1;
    *(v8bf*)&a0       = *(const v8bf*)(aRow0 + kk + kpart);
    *((v8bf*)&a0 + 1) = *(const v8bf*)(aRow0 + kk + kpart + 8);
    *(v8bf*)&a1       = *(const v8bf*)(aRow1 + kk + kpart);
    *((v8bf*)&a1 + 1) = *(const v8bf*)(aRow1 + kk + kpart + 8);
    // B operands: 4 distinct register groups so loads can clause + overlap
    v16bf bv[4];
#pragma unroll
    for (int ct = 0; ct < 4; ct++) {
      int bcol = (cg << 6) + (ct << 4) + lr;    // B column = lane&15
      bv[ct] = *(const v16bf*)(BT + (size_t)bcol * K + kk + (lh << 4));
    }
    if (kk + 32 < K) {
      __builtin_prefetch(aRow0 + kk + 32 + kpart, 0, 3);
      __builtin_prefetch(aRow1 + kk + 32 + kpart, 0, 3);
    }
#pragma unroll
    for (int ct = 0; ct < 4; ct++)
      acc[0][ct] = __builtin_amdgcn_wmma_f32_16x16x32_bf16(
          false, a0, false, bv[ct], (short)0, acc[0][ct], false, false);
#pragma unroll
    for (int ct = 0; ct < 4; ct++)
      acc[1][ct] = __builtin_amdgcn_wmma_f32_16x16x32_bf16(
          false, a1, false, bv[ct], (short)0, acc[1][ct], false, false);
  }

#pragma unroll
  for (int rt = 0; rt < 2; rt++) {
#pragma unroll
    for (int ct = 0; ct < 4; ct++) {
#pragma unroll
      for (int r = 0; r < 8; r++) {
        int m = rowBase + (rt << 4) + r + (lh << 3);   // C row
        int c = (cg << 6) + (ct << 4) + lr;            // C col
        float va = acc[rt][ct][r];
        if (mode == 3) {
          outf[(size_t)m * N + c] += va;
        } else if (mode == 4) {
          float g = 0.5f * va *
                    (1.0f + tanhf(0.7978845608028654f * (va + 0.044715f * va * va * va)));
          outb[(size_t)m * N + c] = (bf16)g;
        } else {
          int bb = m >> 11, s = m & 2047;
          int hh = c >> 5, d = c & 31;
          if (mode == 0)
            outb[(((size_t)bb * Hn + hh) * Sn + s) * DHn + d] =
                (bf16)(va * 0.17677669529663687f);   // 1/sqrt(32)
          else if (mode == 1)
            outb[(((size_t)bb * Hn + hh) * Sn + s) * DHn + d] = (bf16)va;
          else
            outb[(((size_t)bb * Hn + hh) * DHn + d) * Sn + s] = (bf16)va;
        }
      }
    }
  }
}

// ---------------------------------------------------------------------------
// Block-sparse attention. One workgroup (4 waves, 128 thr) per
// (b, h, block n, 16-query subtile). Phase1: 28 WMMA score tiles -> LDS fp32.
// Phase2: fp32 softmax over 448 keys. Phase3: probs @ v_sel via 14 WMMA steps.
// ---------------------------------------------------------------------------
__global__ __launch_bounds__(128) void attn_k(const bf16* __restrict__ q,
                                              const bf16* __restrict__ k,
                                              const bf16* __restrict__ vT,
                                              const int* __restrict__ rand_idx,
                                              bf16* __restrict__ outb) {
  __shared__ __align__(32) float sc[16 * MKn];   // 28 KB scores
  __shared__ __align__(32) bf16  pb[16 * MKn];   // 14 KB probs
  int bx = blockIdx.x;
  int qt = bx & 3;
  int n  = (bx >> 2) & 31;
  int hh = (bx >> 7) & 7;
  int b  = bx >> 10;
  int lane = threadIdx.x & 31;
  int wave = threadIdx.x >> 5;
  int lr = lane & 15, lh = lane >> 4;
  int kpart = lh << 4;

  int idxblk[Mn];
  idxblk[0] = 0;
  idxblk[1] = 1;
  idxblk[2] = (n - 1 < 0) ? 0 : n - 1;
  idxblk[3] = n;
  idxblk[4] = (n + 1 > NBn - 1) ? NBn - 1 : n + 1;
  idxblk[5] = rand_idx[n * 2 + 0];
  idxblk[6] = rand_idx[n * 2 + 1];

  size_t bh = (size_t)b * Hn + hh;
  int s0 = n * BSn + qt * 16;

  // A operand (q tile) stays in registers for all 28 key tiles
  const bf16* qbase = q + (bh * Sn + s0) * DHn;
  v16bf av;
  *(v8bf*)&av       = *(const v8bf*)(qbase + lr * DHn + kpart);
  *((v8bf*)&av + 1) = *(const v8bf*)(qbase + lr * DHn + kpart + 8);

  // Phase 1: scores (q already scaled by 1/sqrt(dh)); 7 key tiles per wave,
  // one key tile (one B load, one WMMA) per loop iteration.
  const bf16* kbase = k + bh * Sn * DHn;
  for (int kt = wave; kt < 28; kt += 4) {
    int blk  = idxblk[kt >> 2];
    int skey = blk * BSn + (kt & 3) * 16 + lr;          // B col = key
    v16bf bv = *(const v16bf*)(kbase + (size_t)skey * DHn + (lh << 4));
    v8f c = (v8f){};
    c = __builtin_amdgcn_wmma_f32_16x16x32_bf16(false, av, false, bv, (short)0,
                                                c, false, false);
#pragma unroll
    for (int r = 0; r < 8; r++)
      sc[(r + (lh << 3)) * MKn + kt * 16 + lr] = c[r];
  }
  __syncthreads();

  // Phase 2: softmax over 448 (one wave per row, 14 values per lane)
  for (int row = wave; row < 16; row += 4) {
    float vv[14];
    float mx = -3.0e38f;
#pragma unroll
    for (int i = 0; i < 14; i++) {
      vv[i] = sc[row * MKn + i * 32 + lane];
      mx = fmaxf(mx, vv[i]);
    }
#pragma unroll
    for (int o = 16; o > 0; o >>= 1) mx = fmaxf(mx, __shfl_xor(mx, o, 32));
    float sm = 0.f;
#pragma unroll
    for (int i = 0; i < 14; i++) { vv[i] = __expf(vv[i] - mx); sm += vv[i]; }
#pragma unroll
    for (int o = 16; o > 0; o >>= 1) sm += __shfl_xor(sm, o, 32);
    float inv = 1.0f / sm;
#pragma unroll
    for (int i = 0; i < 14; i++)
      pb[row * MKn + i * 32 + lane] = (bf16)(vv[i] * inv);
  }
  __syncthreads();

  // Phase 3: o = probs(16x448) @ v_sel(448x32); waves 0,1 own the two dh tiles
  if (wave < 2) {
    int dt = wave;
    const bf16* vbase = vT + (bh * DHn + dt * 16 + lr) * Sn;  // B col = dh
    v8f acc = (v8f){};
    for (int ks = 0; ks < MKn; ks += 32) {
      v16bf pa;
      *(v8bf*)&pa       = *(const v8bf*)(&pb[lr * MKn + ks + kpart]);
      *((v8bf*)&pa + 1) = *(const v8bf*)(&pb[lr * MKn + ks + kpart + 8]);
      int blk = idxblk[ks >> 6];
      v16bf bv = *(const v16bf*)(vbase + blk * BSn + (ks & 63) + (lh << 4));
      acc = __builtin_amdgcn_wmma_f32_16x16x32_bf16(false, pa, false, bv,
                                                    (short)0, acc, false, false);
    }
#pragma unroll
    for (int r = 0; r < 8; r++) {
      int s = s0 + r + (lh << 3);
      outb[((size_t)b * Sn + s) * Dn + hh * DHn + dt * 16 + lr] = (bf16)acc[r];
    }
  }
}

// ---------------------------------------------------------------------------
// Host-side orchestration
// ---------------------------------------------------------------------------
extern "C" void kernel_launch(void* const* d_in, const int* in_sizes, int n_in,
                              void* d_out, int out_size, void* d_ws, size_t ws_size,
                              hipStream_t stream) {
  (void)in_sizes; (void)n_in; (void)out_size; (void)ws_size;
  const long long* ids   = (const long long*)d_in[0];
  const int*   rand_idx  = (const int*)d_in[1];
  const float* emb       = (const float*)d_in[2];
  const float* ln1       = (const float*)d_in[3];
  const float* wq        = (const float*)d_in[4];
  const float* wk        = (const float*)d_in[5];
  const float* wv        = (const float*)d_in[6];
  const float* wo        = (const float*)d_in[7];
  const float* ln2       = (const float*)d_in[8];
  const float* w1        = (const float*)d_in[9];
  const float* w2        = (const float*)d_in[10];
  const float* fln       = (const float*)d_in[11];

  const int rows = Bn * Sn;                      // 16384
  char* w0 = (char*)d_ws;
  float* x    = (float*)(w0);                    // 16,777,216 B
  bf16*  hbf  = (bf16*)(w0 + 16777216);          //  8,388,608 B
  bf16*  qbf  = (bf16*)(w0 + 25165824);          //  8,388,608 B
  bf16*  kbf  = (bf16*)(w0 + 33554432);          //  8,388,608 B
  bf16*  vTb  = (bf16*)(w0 + 41943040);          //  8,388,608 B
  bf16*  attn = (bf16*)(w0 + 50331648);          //  8,388,608 B
  bf16*  ffb  = qbf;                             // alias: q/k/v/attn dead in FFN
  bf16*  wtb  = (bf16*)(w0 + 58720256);          //  3,145,728 B (ends ~61.9 MB)

  // Deterministic per-call: bf16 transposed weights
  for (int l = 0; l < 2; l++) {
    bf16* base = wtb + (size_t)l * 786432;
    wtrans_k<<<(65536 + 255) / 256, 256, 0, stream>>>(wq + (size_t)l * 65536,  base,          256, 256);
    wtrans_k<<<(65536 + 255) / 256, 256, 0, stream>>>(wk + (size_t)l * 65536,  base + 65536,  256, 256);
    wtrans_k<<<(65536 + 255) / 256, 256, 0, stream>>>(wv + (size_t)l * 65536,  base + 131072, 256, 256);
    wtrans_k<<<(65536 + 255) / 256, 256, 0, stream>>>(wo + (size_t)l * 65536,  base + 196608, 256, 256);
    wtrans_k<<<(262144 + 255) / 256, 256, 0, stream>>>(w1 + (size_t)l * 262144, base + 262144, 256, 1024);
    wtrans_k<<<(262144 + 255) / 256, 256, 0, stream>>>(w2 + (size_t)l * 262144, base + 524288, 1024, 256);
  }

  embed_k<<<rows, Dn, 0, stream>>>(ids, emb, x);

  const int gemmBlk256  = ((rows / 32) * (256 / 64)) / 8;   // 256
  const int gemmBlk1024 = ((rows / 32) * (1024 / 64)) / 8;  // 1024

  for (int l = 0; l < 2; l++) {
    bf16* base = wtb + (size_t)l * 786432;
    bf16* wqT = base;
    bf16* wkT = base + 65536;
    bf16* wvT = base + 131072;
    bf16* woT = base + 196608;
    bf16* w1T = base + 262144;
    bf16* w2T = base + 524288;

    rmsnorm_k<bf16><<<rows / 8, 256, 0, stream>>>(x, ln1 + l * Dn, hbf, rows);

    gemm_bf16_k<<<gemmBlk256, 256, 0, stream>>>(hbf, wqT, rows, 256, 256, 0, qbf, nullptr);
    gemm_bf16_k<<<gemmBlk256, 256, 0, stream>>>(hbf, wkT, rows, 256, 256, 1, kbf, nullptr);
    gemm_bf16_k<<<gemmBlk256, 256, 0, stream>>>(hbf, wvT, rows, 256, 256, 2, vTb, nullptr);

    attn_k<<<Bn * Hn * NBn * 4, 128, 0, stream>>>(qbf, kbf, vTb, rand_idx, attn);

    gemm_bf16_k<<<gemmBlk256, 256, 0, stream>>>(attn, woT, rows, 256, 256, 3, nullptr, x);

    rmsnorm_k<bf16><<<rows / 8, 256, 0, stream>>>(x, ln2 + l * Dn, hbf, rows);

    gemm_bf16_k<<<gemmBlk1024, 256, 0, stream>>>(hbf, w1T, rows, 256, 1024, 4, ffb, nullptr);
    gemm_bf16_k<<<gemmBlk256, 256, 0, stream>>>(ffb, w2T, rows, 1024, 256, 3, nullptr, x);
  }

  rmsnorm_k<float><<<rows / 8, 256, 0, stream>>>(x, fln, (float*)d_out, rows);
}
